// BiDAFSelfAttention_26697516712116
// MI455X (gfx1250) — matrix operand order repaired and checked by reading, not compiled, not run on hardware
//
#include <hip/hip_runtime.h>

#define B_  4
#define T_  2048
#define H_  1024
#define NH_ 16
#define DK_ 64
#define M_  (B_*T_)   // 8192

typedef __attribute__((ext_vector_type(16))) __bf16 v16bf;
typedef __attribute__((ext_vector_type(2)))  __bf16 v2bf;
typedef __attribute__((ext_vector_type(8)))  float  v8f;

union AF { v16bf v; uint4 q[2]; unsigned short u[16]; };
union CF { v8f v; float f[8]; };

__device__ inline unsigned short f2bf(float f) {
  unsigned int u = __float_as_uint(f);
  u += 0x7FFFu + ((u >> 16) & 1u);   // RNE
  return (unsigned short)(u >> 16);
}
__device__ inline unsigned pk2bf(float a, float b) {
#if __has_builtin(__builtin_amdgcn_cvt_pk_bf16_f32)
  v2bf r = __builtin_amdgcn_cvt_pk_bf16_f32(a, b);
  unsigned out; __builtin_memcpy(&out, &r, 4); return out;
#else
  return (unsigned)f2bf(a) | ((unsigned)f2bf(b) << 16);
#endif
}

// gfx1250 async copy global -> LDS (ASYNCcnt), 16B per lane
__device__ inline void async_ld_b128(unsigned lds_addr, unsigned long long gaddr) {
  asm volatile("global_load_async_to_lds_b128 %0, %1, off"
               :: "v"(lds_addr), "v"(gaddr) : "memory");
}
__device__ inline void wait_async0() {
  asm volatile("s_wait_asynccnt 0x0" ::: "memory");
}

// ---------------------------------------------------------------------------
// Y[M,1024] = X[M,1024] @ W^T + bias.  MODE 0/1/2: bf16 head-split out
// (MODE 0 folds 1/sqrt(DK)); MODE 3: fp32 flat to d_out.
// 128x128x32 tiles, LDS double-buffered, 1 barrier/iter.
// ---------------------------------------------------------------------------
#define LDA 40
#define LDB 136

template<int MODE>
__global__ __launch_bounds__(256)
void proj_gemm(const float* __restrict__ X, const float* __restrict__ W,
               const float* __restrict__ bias,
               unsigned short* __restrict__ outb, float* __restrict__ outf)
{
  __shared__ unsigned short As[2][128 * LDA];
  __shared__ unsigned short Bs[2][32 * LDB];

  const int tid  = threadIdx.x;
  const int lane = tid & 31;
  const int w    = tid >> 5;
  const int wm   = w >> 1;
  const int wn   = w & 1;
  const int nblk = blockIdx.x * 128;
  const int mblk = blockIdx.y * 128;
  const int l16  = lane & 15;
  const int hi   = lane >> 4;
  const int klo  = hi * 8;

  float4 ra[4], rb[4];
  // prologue: stage k-tile 0 into buffer 0
  #pragma unroll
  for (int i = 0; i < 4; ++i) {
    int lin = tid + i * 256, row = lin >> 3, k4 = (lin & 7) * 4;
    ra[i] = *(const float4*)(X + (size_t)(mblk + row) * H_ + k4);
    rb[i] = *(const float4*)(W + (size_t)(nblk + row) * H_ + k4);
  }
  #pragma unroll
  for (int i = 0; i < 4; ++i) {
    int lin = tid + i * 256, row = lin >> 3, k4 = (lin & 7) * 4;
    *(unsigned*)(&As[0][row * LDA + k4])     = pk2bf(ra[i].x, ra[i].y);
    *(unsigned*)(&As[0][row * LDA + k4 + 2]) = pk2bf(ra[i].z, ra[i].w);
    Bs[0][(k4 + 0) * LDB + row] = f2bf(rb[i].x);
    Bs[0][(k4 + 1) * LDB + row] = f2bf(rb[i].y);
    Bs[0][(k4 + 2) * LDB + row] = f2bf(rb[i].z);
    Bs[0][(k4 + 3) * LDB + row] = f2bf(rb[i].w);
  }
  __syncthreads();

  CF acc[2][4];
  #pragma unroll
  for (int i = 0; i < 2; ++i)
    #pragma unroll
    for (int t = 0; t < 4; ++t)
      #pragma unroll
      for (int j = 0; j < 8; ++j) acc[i][t].f[j] = 0.0f;

  for (int kt = 0; kt < H_ / 32; ++kt) {
    const int cur = kt & 1;
    if (kt < 31) {           // prefetch next tile into registers (overlaps WMMA)
      const int k0 = (kt + 1) * 32;
      #pragma unroll
      for (int i = 0; i < 4; ++i) {
        int lin = tid + i * 256, row = lin >> 3, k4 = (lin & 7) * 4;
        ra[i] = *(const float4*)(X + (size_t)(mblk + row) * H_ + k0 + k4);
        rb[i] = *(const float4*)(W + (size_t)(nblk + row) * H_ + k0 + k4);
      }
    }
    AF a[2];
    #pragma unroll
    for (int i = 0; i < 2; ++i) {
      int row = wm * 32 + i * 16 + l16;
      a[i].q[0] = *(const uint4*)(&As[cur][row * LDA + klo]);
      a[i].q[1] = *(const uint4*)(&As[cur][row * LDA + 16 + klo]);
    }
    #pragma unroll
    for (int t = 0; t < 4; ++t) {
      AF bfr;
      int n = wn * 64 + t * 16;
      #pragma unroll
      for (int j = 0; j < 8; ++j) {
        bfr.u[j]     = Bs[cur][(klo + j) * LDB + n + l16];
        bfr.u[8 + j] = Bs[cur][(16 + klo + j) * LDB + n + l16];
      }
      #pragma unroll
      for (int i = 0; i < 2; ++i)
        acc[i][t].v = __builtin_amdgcn_wmma_f32_16x16x32_bf16(
            false, a[i].v, false, bfr.v, (short)0, acc[i][t].v, false, false);
    }
    if (kt < 31) {
      #pragma unroll
      for (int i = 0; i < 4; ++i) {
        int lin = tid + i * 256, row = lin >> 3, k4 = (lin & 7) * 4;
        *(unsigned*)(&As[1 - cur][row * LDA + k4])     = pk2bf(ra[i].x, ra[i].y);
        *(unsigned*)(&As[1 - cur][row * LDA + k4 + 2]) = pk2bf(ra[i].z, ra[i].w);
        Bs[1 - cur][(k4 + 0) * LDB + row] = f2bf(rb[i].x);
        Bs[1 - cur][(k4 + 1) * LDB + row] = f2bf(rb[i].y);
        Bs[1 - cur][(k4 + 2) * LDB + row] = f2bf(rb[i].z);
        Bs[1 - cur][(k4 + 3) * LDB + row] = f2bf(rb[i].w);
      }
    }
    __syncthreads();
  }

  #pragma unroll
  for (int i = 0; i < 2; ++i)
    #pragma unroll
    for (int t = 0; t < 4; ++t) {
      int n = nblk + wn * 64 + t * 16 + l16;
      float bv = bias[n];
      #pragma unroll
      for (int r = 0; r < 8; ++r) {
        int m = mblk + wm * 32 + i * 16 + r + hi * 8;
        if constexpr (MODE < 3) {
          float val = acc[i][t].f[r] + bv;
          if constexpr (MODE == 0) val *= 0.125f;
          int bb = m >> 11, tq = m & (T_ - 1);
          int hh = n >> 6,  dd = n & 63;
          outb[(((size_t)(bb * NH_ + hh)) * T_ + tq) * DK_ + dd] = f2bf(val);
        } else {
          outf[(size_t)m * H_ + n] = acc[i][t].f[r] + bv;
        }
      }
    }
}

// ---------------------------------------------------------------------------
// Flash attention, S^T = K*Q^T formulation:
//  - K,V chunks are row-major in LDS -> pure async copies (ASYNCcnt)
//  - Q transposed once per WG into LDS, B-fragments resident in regs
//  - softmax: per-lane q column, 2 cross-lane shuffles per chunk
// ---------------------------------------------------------------------------
#define LDKV 72
#define LDQ  136
#define LDP  40

__global__ __launch_bounds__(256)
void flash_attn(const unsigned short* __restrict__ Q,
                const unsigned short* __restrict__ K,
                const unsigned short* __restrict__ V,
                const unsigned char* __restrict__ mask,
                float* __restrict__ ctx)
{
  __shared__ __align__(16) unsigned char smem[54272];
  unsigned short* Ks = (unsigned short*)smem;               // [128][LDKV]
  unsigned short* Vs = (unsigned short*)(smem + 18432);     // [128][LDKV]
  unsigned short* Qt = (unsigned short*)(smem + 36864);     // [64][LDQ]   (start only)
  unsigned short* Ps = (unsigned short*)(smem + 36864);     // 8 x [16][LDP] (overlaps Qt)

  const int tid  = threadIdx.x;
  const int lane = tid & 31;
  const int w    = tid >> 5;
  const int l16  = lane & 15;
  const int hi   = lane >> 4;
  const int klo  = hi * 8;

  const int qblk = blockIdx.x * 128;
  const int h    = blockIdx.y;
  const int b    = blockIdx.z;
  const size_t bh = ((size_t)(b * NH_ + h)) * T_ * DK_;
  const unsigned char* mb = mask + (size_t)b * T_;

  const unsigned ksB = (unsigned)(unsigned long long)Ks;    // LDS addr = low 32 bits
  const unsigned vsB = (unsigned)(unsigned long long)Vs;

  // kick off async copy of chunk 0 while we transpose Q
  #pragma unroll
  for (int i = 0; i < 4; ++i) {
    int lin = tid + i * 256, key = lin >> 3, d8 = (lin & 7) * 8;
    unsigned loff = (unsigned)(key * LDKV + d8) * 2u;
    async_ld_b128(ksB + loff, (unsigned long long)(K + bh + (size_t)key * DK_ + d8));
    async_ld_b128(vsB + loff, (unsigned long long)(V + bh + (size_t)key * DK_ + d8));
  }

  // stage Q^T [d][q] once
  #pragma unroll
  for (int i = 0; i < 8; ++i) {
    int lin = tid + i * 256, qr = lin >> 4, d4 = (lin & 15) * 4;
    ushort4 qv = *(const ushort4*)(Q + bh + (size_t)(qblk + qr) * DK_ + d4);
    Qt[(d4 + 0) * LDQ + qr] = qv.x;
    Qt[(d4 + 1) * LDQ + qr] = qv.y;
    Qt[(d4 + 2) * LDQ + qr] = qv.z;
    Qt[(d4 + 3) * LDQ + qr] = qv.w;
  }
  __syncthreads();

  AF qb[2];                       // Q^T B-fragments, resident all kernel
  #pragma unroll
  for (int dt = 0; dt < 2; ++dt) {
    #pragma unroll
    for (int j = 0; j < 8; ++j) {
      qb[dt].u[j]     = Qt[(dt * 32 + klo + j) * LDQ + w * 16 + l16];
      qb[dt].u[8 + j] = Qt[(dt * 32 + 16 + klo + j) * LDQ + w * 16 + l16];
    }
  }
  __syncthreads();                // Qt region now reusable as Ps

  CF acc[4];
  #pragma unroll
  for (int t = 0; t < 4; ++t)
    #pragma unroll
    for (int j = 0; j < 8; ++j) acc[t].f[j] = 0.0f;
  float mrow = -1e30f, lrow = 0.0f;

  for (int c = 0; c < T_ / 128; ++c) {
    wait_async0();
    __syncthreads();

    // S^T = K * Q^T : rows = keys, cols = q (8 key-tiles x 2 k-steps)
    CF s[8];
    #pragma unroll
    for (int f = 0; f < 8; ++f) {
      #pragma unroll
      for (int j = 0; j < 8; ++j) s[f].f[j] = 0.0f;
      #pragma unroll
      for (int dt = 0; dt < 2; ++dt) {
        AF ak;
        const unsigned short* kr = &Ks[(f * 16 + l16) * LDKV + dt * 32 + klo];
        ak.q[0] = *(const uint4*)(kr);
        ak.q[1] = *(const uint4*)(kr + 16);
        s[f].v = __builtin_amdgcn_wmma_f32_16x16x32_bf16(
            false, ak.v, false, qb[dt].v, (short)0, s[f].v, false, false);
      }
      // PAD mask: keys f*16 + 8*hi + [0..7]; skip the whole fixup when no PAD
      unsigned long long mm = *(const unsigned long long*)(mb + c * 128 + f * 16 + hi * 8);
      if (__any(mm != 0ull)) {
        unsigned mml = (unsigned)mm, mmh = (unsigned)(mm >> 32);
        #pragma unroll
        for (int r = 0; r < 4; ++r)
          if ((mml >> (8 * r)) & 0xffu) s[f].f[r] = -1e9f;
        #pragma unroll
        for (int r = 4; r < 8; ++r)
          if ((mmh >> (8 * (r - 4))) & 0xffu) s[f].f[r] = -1e9f;
      }
    }

    // online softmax: this lane owns one q column; other half of keys in lane^16
    float vmax = -1e30f;
    #pragma unroll
    for (int f = 0; f < 8; ++f)
      #pragma unroll
      for (int r = 0; r < 8; ++r) vmax = fmaxf(vmax, s[f].f[r]);
    vmax = fmaxf(vmax, __shfl_xor(vmax, 16, 32));
    float newm  = fmaxf(mrow, vmax);
    float alpha = __expf(mrow - newm);
    mrow = newm;
    float rs = 0.0f;
    #pragma unroll
    for (int f = 0; f < 8; ++f)
      #pragma unroll
      for (int r = 0; r < 8; ++r) { s[f].f[r] = __expf(s[f].f[r] - newm); rs += s[f].f[r]; }
    rs += __shfl_xor(rs, 16, 32);
    lrow = lrow * alpha + rs;

    float alr[8];                 // redistribute alpha to C-layout rows
    #pragma unroll
    for (int r = 0; r < 8; ++r) alr[r] = __shfl(alpha, r + 8 * hi, 16);
    #pragma unroll
    for (int t = 0; t < 4; ++t)
      #pragma unroll
      for (int r = 0; r < 8; ++r) acc[t].f[r] *= alr[r];

    // ctx += P V : write P^T (D-layout) transposed into per-wave LDS, reload A-layout
    unsigned short* Pw = Ps + w * (16 * LDP);
    #pragma unroll
    for (int kt = 0; kt < 4; ++kt) {
      #pragma unroll
      for (int fo = 0; fo < 2; ++fo) {
        int f = kt * 2 + fo;
        #pragma unroll
        for (int r = 0; r < 8; r += 2)
          *(unsigned*)(&Pw[l16 * LDP + fo * 16 + 8 * hi + r]) =
              pk2bf(s[f].f[r], s[f].f[r + 1]);
      }
      AF ap;                      // same-wave LDS ops are in-order
      ap.q[0] = *(const uint4*)(&Pw[l16 * LDP + klo]);
      ap.q[1] = *(const uint4*)(&Pw[l16 * LDP + 16 + klo]);
      #pragma unroll
      for (int t = 0; t < 4; ++t) {
        AF bv;
        const unsigned short* vr = &Vs[(kt * 32 + klo) * LDKV + t * 16 + l16];
        #pragma unroll
        for (int j = 0; j < 8; ++j) {
          bv.u[j]     = vr[j * LDKV];
          bv.u[8 + j] = vr[(16 + j) * LDKV];
        }
        acc[t].v = __builtin_amdgcn_wmma_f32_16x16x32_bf16(
            false, ap.v, false, bv.v, (short)0, acc[t].v, false, false);
      }
    }
    __syncthreads();              // all waves done with Ks/Vs

    if (c + 1 < T_ / 128) {       // issue next chunk's async copy
      const unsigned short* kp = K + bh + (size_t)(c + 1) * 128 * DK_;
      const unsigned short* vp = V + bh + (size_t)(c + 1) * 128 * DK_;
      #pragma unroll
      for (int i = 0; i < 4; ++i) {
        int lin = tid + i * 256, key = lin >> 3, d8 = (lin & 7) * 8;
        unsigned loff = (unsigned)(key * LDKV + d8) * 2u;
        async_ld_b128(ksB + loff, (unsigned long long)(kp + (size_t)key * DK_ + d8));
        async_ld_b128(vsB + loff, (unsigned long long)(vp + (size_t)key * DK_ + d8));
      }
    }
  }

  float lr[8];
  #pragma unroll
  for (int r = 0; r < 8; ++r) lr[r] = __shfl(lrow, r + 8 * hi, 16);
  #pragma unroll
  for (int t = 0; t < 4; ++t)
    #pragma unroll
    for (int r = 0; r < 8; ++r) {
      int row = qblk + w * 16 + r + 8 * hi;
      int col = h * DK_ + t * 16 + l16;
      ctx[((size_t)(b * T_ + row)) * H_ + col] = acc[t].f[r] / lr[r];
    }
}

// ---------------------------------------------------------------------------
extern "C" void kernel_launch(void* const* d_in, const int* in_sizes, int n_in,
                              void* d_out, int out_size, void* d_ws, size_t ws_size,
                              hipStream_t stream) {
  (void)in_sizes; (void)n_in; (void)out_size; (void)ws_size;
  const float* q  = (const float*)d_in[0];
  const float* k  = (const float*)d_in[1];
  const float* v  = (const float*)d_in[2];
  const unsigned char* mask = (const unsigned char*)d_in[3];
  const float* Wq = (const float*)d_in[4];
  const float* bq = (const float*)d_in[5];
  const float* Wk = (const float*)d_in[6];
  const float* bk = (const float*)d_in[7];
  const float* Wv = (const float*)d_in[8];
  const float* bv = (const float*)d_in[9];
  const float* Wo = (const float*)d_in[10];
  const float* bo = (const float*)d_in[11];

  const size_t QKV = (size_t)B_ * NH_ * T_ * DK_;
  unsigned short* qbf = (unsigned short*)d_ws;   // bf16 [B,NH,T,DK]
  unsigned short* kbf = qbf + QKV;
  unsigned short* vbf = kbf + QKV;
  float* ctx = (float*)(vbf + QKV);              // fp32 [B,T,H]

  dim3 ggrid(H_ / 128, M_ / 128);
  proj_gemm<0><<<ggrid, 256, 0, stream>>>(q, Wq, bq, qbf, nullptr);
  proj_gemm<1><<<ggrid, 256, 0, stream>>>(k, Wk, bk, kbf, nullptr);
  proj_gemm<2><<<ggrid, 256, 0, stream>>>(v, Wv, bv, vbf, nullptr);
  flash_attn<<<dim3(T_ / 128, NH_, B_), 256, 0, stream>>>(qbf, kbf, vbf, mask, ctx);
  proj_gemm<3><<<ggrid, 256, 0, stream>>>(ctx, Wo, bo, nullptr, (float*)d_out);
}
